// DirectionalConvLayer_7567732375855
// MI455X (gfx1250) — compile-verified, matrix operand order checked by reading, not run on hardware
//
#include <hip/hip_runtime.h>
#include <hip/hip_bf16.h>

typedef __attribute__((ext_vector_type(16))) _Float16 v16h;
typedef __attribute__((ext_vector_type(8)))  _Float16 v8h;
typedef __attribute__((ext_vector_type(8)))  float    v8f;

#define CCH   64      // channels
#define WWID  256     // row width
#define HHGT  256     // number of rows
#define KTOT  192     // K = 64 ci * 3 taps (k = kw*64 + ci)
#define ROWP  258     // padded row width (+1 halo each side)
#define EPS   1e-5f

union FragB { v16h v; v8h h[2]; };

// One directional scan (forward==1: top-down over rows 0..H-1 reading x;
// forward==0: bottom-up reading/overwriting the f-stack in `out` in place).
// One block per batch element n; 256 threads = 8 waves.
__global__ __launch_bounds__(256) void dirconv_scan(
    const float* __restrict__ Wfull,   // (64,64,3,3) OIHW
    const float* __restrict__ bias,    // (64,)
    const float* __restrict__ src,     // residual source rows (x or f-stack)
    float*       __restrict__ out,     // f-stack (fwd) / final output (bwd)
    int forward)
{
    // Weights co-major: wA[co*KTOT + k]  -> consecutive k contiguous.
    __shared__ _Float16 wA[CCH * KTOT];       // 24576 B
    // State row w-major with halo: row16T[w*CCH + ci], w in 0..257.
    __shared__ _Float16 row16T[ROWP * CCH];   // 33024 B
    __shared__ float    yrow[CCH * WWID];     // 65536 B
    __shared__ float    redS [CCH][4];
    __shared__ float    redS2[CCH][4];
    __shared__ float    meanA[CCH];
    __shared__ float    rstdA[CCH];
    __shared__ float    bvals[CCH];

    const int tid  = threadIdx.x;
    const int lane = tid & 31;
    const int wv   = tid >> 5;
    const int n    = blockIdx.x;

    // ---- one-time setup: weights (middle kh row only), bias, halo zeros ----
    for (int e = tid; e < CCH * KTOT; e += 256) {
        int co = e / KTOT;
        int k  = e % KTOT;        // k = kw*64 + ci
        int kw = k >> 6;
        int ci = k & 63;
        wA[e] = (_Float16)Wfull[(co * 64 + ci) * 9 + 3 + kw];  // W[co][ci][1][kw]
    }
    if (tid < CCH) {
        bvals[tid] = bias[tid];
        row16T[0 * CCH + tid]          = (_Float16)0.f;  // left zero-pad (w = -1)
        row16T[(ROWP - 1) * CCH + tid] = (_Float16)0.f;  // right zero-pad (w = 256)
    }
    __syncthreads();

    // ---- per-wave constant A fragments (weights), 6 k-steps of 32 ----
    // Fragment half layout (16-bit 16x32, CDNA5 ISA 7.12.2):
    //   halves 0..7  = K kbase+0..7,  halves 8..15 = K kbase+16..23,
    //   kbase = 8 for lanes 16..31 else 0.  Two contiguous 16B runs.
    const int ct    = wv >> 1;          // co-tile 0..3 (16 channels each)
    const int whalf = wv & 1;           // which 8 w-tiles this wave owns
    const int coL   = ct * 16 + (lane & 15);
    const int kbase = (lane & 16) ? 8 : 0;
    FragB afrag[6];
#pragma unroll
    for (int ks = 0; ks < 6; ++ks) {
        const _Float16* ap = &wA[coL * KTOT + ks * 32 + kbase];
        afrag[ks].h[0] = *(const v8h*)(ap);
        afrag[ks].h[1] = *(const v8h*)(ap + 16);
    }

    // ---- initial state: f[0] = x[0] (fwd) / g[H-1] = f[H-1] (bwd) ----
    const int row0 = forward ? 0 : (HHGT - 1);
    for (int e = tid; e < CCH * WWID; e += 256) {
        int co = e >> 8, w = e & 255;
        size_t gi = ((size_t)(n * CCH + co) * HHGT + row0) * WWID + w;
        float f = src[gi];
        if (forward) out[gi] = f;       // bwd: src==out, row already holds f
        row16T[(w + 1) * CCH + co] = (_Float16)f;
    }
    __syncthreads();

    // ---- sequential scan over rows ----
    for (int step = 1; step < HHGT; ++step) {
        const int row = forward ? step : (HHGT - 1 - step);

        // GEMM: y[co, w] = sum_{ci,kw} state[ci, w+kw-1] * W1[co, ci, kw]
        // Each wave: 8 w-tiles x 6 WMMA k-steps. Uniform over all threads.
#pragma unroll 1
        for (int t8 = 0; t8 < 8; ++t8) {
            const int w0  = (whalf * 8 + t8) * 16;
            const int col = w0 + (lane & 15);
            v8f acc = {0.f, 0.f, 0.f, 0.f, 0.f, 0.f, 0.f, 0.f};
#pragma unroll
            for (int ks = 0; ks < 6; ++ks) {
                const int kw     = ks >> 1;         // tap for this k-step
                const int cibase = (ks & 1) * 32;   // ci range start
                // padded w index: (col + kw - 1) + 1 = col + kw
                const _Float16* bp =
                    &row16T[(size_t)(col + kw) * CCH + cibase + kbase];
                FragB bf;
                bf.h[0] = *(const v8h*)(bp);        // K kbase+0..7
                bf.h[1] = *(const v8h*)(bp + 16);   // K kbase+16..23
                acc = __builtin_amdgcn_wmma_f32_16x16x32_f16(
                    false, afrag[ks].v, false, bf.v, (short)0, acc,
                    false, false);
            }
            const int coBase = ct * 16 + ((lane & 16) ? 8 : 0);
#pragma unroll
            for (int r = 0; r < 8; ++r) {
                int co = coBase + r;
                yrow[co * WWID + col] = acc[r] + bvals[co];
            }
        }
        __syncthreads();

        // Per-channel mean / biased var over the 256-wide row (2-level LDS).
        {
            int co = tid >> 2, q = tid & 3;
            const float* yp = &yrow[co * WWID + q * 64];
            float s = 0.f, s2 = 0.f;
#pragma unroll 8
            for (int i = 0; i < 64; ++i) { float v = yp[i]; s += v; s2 += v * v; }
            redS[co][q] = s; redS2[co][q] = s2;
        }
        __syncthreads();
        if (tid < CCH) {
            float s  = redS [tid][0] + redS [tid][1] + redS [tid][2] + redS [tid][3];
            float s2 = redS2[tid][0] + redS2[tid][1] + redS2[tid][2] + redS2[tid][3];
            float m   = s * (1.0f / WWID);
            float var = s2 * (1.0f / WWID) - m * m;
            meanA[tid] = m;
            rstdA[tid] = rsqrtf(var + EPS);
        }
        __syncthreads();

        // Normalize + ELU + residual; update f16 state + f32 global stack.
        for (int e = tid; e < CCH * WWID; e += 256) {
            int co = e >> 8, w = e & 255;
            float v = (yrow[e] - meanA[co]) * rstdA[co];
            v = v > 0.f ? v : expm1f(v);
            size_t gi = ((size_t)(n * CCH + co) * HHGT + row) * WWID + w;
            float f = v + src[gi];      // bwd: reads f[p] before overwriting
            out[gi] = f;
            row16T[(w + 1) * CCH + co] = (_Float16)f;
        }
        __syncthreads();
    }
}

extern "C" void kernel_launch(void* const* d_in, const int* in_sizes, int n_in,
                              void* d_out, int out_size, void* d_ws, size_t ws_size,
                              hipStream_t stream) {
    const float* x = (const float*)d_in[0];   // (8,64,256,256)
    const float* W = (const float*)d_in[1];   // (64,64,3,3)
    const float* b = (const float*)d_in[2];   // (64,)
    float* out = (float*)d_out;

    // Forward scan: writes the f-stack into d_out.
    dirconv_scan<<<8, 256, 0, stream>>>(W, b, x, out, 1);
    // Backward scan: reads f[p] from d_out, overwrites with g[p] in place.
    dirconv_scan<<<8, 256, 0, stream>>>(W, b, out, out, 0);
}
